// MultiHeadAttention_71305047048557
// MI455X (gfx1250) — compile-verified, matrix operand order checked by reading, not run on hardware
//
#include <hip/hip_runtime.h>
#include <hip/hip_bf16.h>

// MHA forward for MI455X (gfx1250, wave32).
// Compute-bound (137 GFLOP vs ~100MB traffic @ 23.3 TB/s) -> bf16 WMMA
// (V_WMMA_F32_16X16X32_BF16; fp32 WMMA is only 16x16x4). All bf16
// intermediates (~50MB) fit in the 192MB L2.
// This revision: ping-pong (unroll-by-2) GEMM k-loop so prefetched fragments
// land directly in their consuming registers -> removes the v_dual_mov_b32
// rotation copies seen in the previous build.

#define B_  2
#define L_  2048
#define D_  1024
#define H_  16
#define DH_ 64

typedef __attribute__((ext_vector_type(16))) __bf16 v16bf;
typedef __attribute__((ext_vector_type(8)))  float  v8f;

union Frag16 { v16bf v; uint4 q[2]; };

// A-fragment (16x32 bf16, row-major source, leading dim `ld`): lane holds row
// `lm`, K-halves split by hf (lane/16); VGPRs 0-3 = K hf*8+0..7, VGPRs 4-7 =
// K hf*8+16..23 -> two contiguous b128 loads per lane. B-fragments (32x16,
// column-major per ISA) use the same indexing on the transposed operand.
__device__ __forceinline__ v16bf load_frag(const __bf16* p, int row, int ld,
                                           int k0, int hf) {
  Frag16 f;
  const __bf16* rp = p + (size_t)row * ld + k0 + hf * 8;
  f.q[0] = *(const uint4*)(rp);
  f.q[1] = *(const uint4*)(rp + 16);
  return f.v;
}

__device__ __forceinline__ v8f wmma_bf16(v16bf a, v16bf b, v8f c) {
  return __builtin_amdgcn_wmma_f32_16x16x32_bf16(
      false, a, false, b, (short)0, c, false, false);
}

__device__ __forceinline__ float half_max16(float v) {
#pragma unroll
  for (int m = 1; m < 16; m <<= 1) v = fmaxf(v, __shfl_xor(v, m, 32));
  return v;
}
__device__ __forceinline__ float half_sum16(float v) {
#pragma unroll
  for (int m = 1; m < 16; m <<= 1) v += __shfl_xor(v, m, 32);
  return v;
}

// ---------------- converts ----------------
__global__ void cvt_bf16(const float* __restrict__ in, __bf16* __restrict__ out,
                         int n) {
  int i = blockIdx.x * blockDim.x + threadIdx.x;
  if (i < n) out[i] = (__bf16)in[i];
}

__global__ void cvt_transpose(const float* __restrict__ w,
                              __bf16* __restrict__ wT) {
  int i = blockIdx.x * blockDim.x + threadIdx.x;  // wT index: n*D + k
  int n = i >> 10, k = i & (D_ - 1);
  wT[i] = (__bf16)w[(size_t)k * D_ + n];
}

// ---------------- GEMM helpers ----------------------------------------------
struct Frags {
  v16bf a0, a1, b[4];
};

__device__ __forceinline__ Frags load_all(const __bf16* __restrict__ A,
                                          const __bf16* __restrict__ Bt,
                                          int m0, int n0, int lm, int hf,
                                          int k0) {
  Frags t;
  t.a0 = load_frag(A, m0 + lm, D_, k0, hf);
  t.a1 = load_frag(A, m0 + 16 + lm, D_, k0, hf);
#pragma unroll
  for (int f = 0; f < 4; ++f)
    t.b[f] = load_frag(Bt, n0 + f * 16 + lm, D_, k0, hf);
  return t;
}

__device__ __forceinline__ void compute8(v8f acc[8], const Frags& t) {
#pragma unroll
  for (int f = 0; f < 4; ++f) {
    acc[f]     = wmma_bf16(t.a0, t.b[f], acc[f]);
    acc[4 + f] = wmma_bf16(t.a1, t.b[f], acc[4 + f]);
  }
}

// ---------------- GEMM: C[4096,1024] = A * Bt^T + bias ----------------------
// Wave tile 32(M) x 64(N): 8 accumulators; each B-fragment reused by 2 A rows.
// Ping-pong k-loop: buffer f0/f1 alternate so loads overlap the 8 WMMAs with
// no register rotation. mode 0: bf16 [B,H,L,Dh]; 2: bf16 [B,H,Dh,L]; 3: fp32.
__global__ void __launch_bounds__(128) gemm_bf16(
    const __bf16* __restrict__ A, const __bf16* __restrict__ Bt,
    const float* __restrict__ bias, void* __restrict__ out, int mode) {
  const int lane = threadIdx.x & 31;
  const int wave = threadIdx.x >> 5;
  const int hf = lane >> 4;
  const int lm = lane & 15;
  const int m0 = (blockIdx.x * 4 + wave) * 32;
  const int n0 = blockIdx.y * 64;

  v8f acc[8] = {};
  Frags f0 = load_all(A, Bt, m0, n0, lm, hf, 0);
  Frags f1 = load_all(A, Bt, m0, n0, lm, hf, 32);
#pragma unroll 1
  for (int k0 = 0; k0 < D_ - 64; k0 += 64) {
    compute8(acc, f0);
    f0 = load_all(A, Bt, m0, n0, lm, hf, k0 + 64);
    compute8(acc, f1);
    f1 = load_all(A, Bt, m0, n0, lm, hf, k0 + 96);
  }
  compute8(acc, f0);
  compute8(acc, f1);

#pragma unroll
  for (int mm = 0; mm < 2; ++mm) {
#pragma unroll
    for (int f = 0; f < 4; ++f) {
      const int n = n0 + f * 16 + lm;
      const float bv = bias[n];
#pragma unroll
      for (int r = 0; r < 8; ++r) {
        const int m = m0 + mm * 16 + r + hf * 8;
        const float val = acc[mm * 4 + f][r] + bv;
        if (mode == 3) {
          ((float*)out)[(size_t)m * D_ + n] = val;
        } else {
          const int bidx = m >> 11, l = m & (L_ - 1);
          const int h = n >> 6, d = n & (DH_ - 1);
          __bf16* ob = (__bf16*)out;
          if (mode == 2)
            ob[((size_t)(bidx * H_ + h) * DH_ + d) * L_ + l] = (__bf16)val;
          else
            ob[((size_t)(bidx * H_ + h) * L_ + l) * DH_ + d] = (__bf16)val;
        }
      }
    }
  }
}

// ---------------- flash attention -------------------------------------------
// One wave per 32-row q-tile (two 16-row WMMA tiles sharing K/V fragments);
// online softmax in log2 domain; 32-key blocks -> 16 WMMAs per block. K/V
// fragment loads are hoisted ahead of the softmax VALU chain for overlap.
__global__ void __launch_bounds__(128) flash_kernel(
    const __bf16* __restrict__ Q, const __bf16* __restrict__ K,
    const __bf16* __restrict__ Vt, __bf16* __restrict__ AO) {
  __shared__ alignas(16) __bf16 lds[4][32 * 32];  // per-wave P tile (2KB)
  const int lane = threadIdx.x & 31;
  const int wave = threadIdx.x >> 5;
  const int hf = lane >> 4;
  const int lm = lane & 15;
  const int tile = blockIdx.x * 4 + wave;          // 0..2047
  const int qt = tile & (L_ / 32 - 1);
  const int bh = tile >> 6;                        // b*H + h
  const int q0 = qt * 32;
  const __bf16* Qp = Q + (size_t)bh * L_ * DH_;
  const __bf16* Kp = K + (size_t)bh * L_ * DH_;
  const __bf16* Vp = Vt + (size_t)bh * DH_ * L_;   // [Dh, L]

  v16bf qa[2][2];
#pragma unroll
  for (int mm = 0; mm < 2; ++mm) {
    qa[mm][0] = load_frag(Qp, q0 + mm * 16 + lm, DH_, 0, hf);
    qa[mm][1] = load_frag(Qp, q0 + mm * 16 + lm, DH_, 32, hf);
  }

  v8f o[2][4] = {};
  float mrun[2][8], lrun[2][8];
#pragma unroll
  for (int mm = 0; mm < 2; ++mm)
#pragma unroll
    for (int r = 0; r < 8; ++r) {
      mrun[mm][r] = -__builtin_inff();
      lrun[mm][r] = 0.f;
    }
  const float cscale = 1.4426950408889634f / 8.0f;  // log2(e)/sqrt(Dh)
  __bf16* pl = lds[wave];

  const int nblk = (q0 >> 5) + 1;  // causal: key blocks covering keys <= q0+31
#pragma unroll 1
  for (int jb = 0; jb < nblk; ++jb) {
    const int kb0 = jb * 32;
    // K fragments (shared by both q row-tiles).
    v16bf kf[2][2];
#pragma unroll
    for (int f = 0; f < 2; ++f) {
      kf[f][0] = load_frag(Kp, kb0 + f * 16 + lm, DH_, 0, hf);
      kf[f][1] = load_frag(Kp, kb0 + f * 16 + lm, DH_, 32, hf);
    }
    // V fragments hoisted: their loads overlap the softmax VALU chain.
    v16bf vb[4];
#pragma unroll
    for (int f = 0; f < 4; ++f)
      vb[f] = load_frag(Vp, f * 16 + lm, L_, kb0, hf);

    v8f s[2][2] = {};
#pragma unroll
    for (int mm = 0; mm < 2; ++mm)
#pragma unroll
      for (int f = 0; f < 2; ++f) {
        s[mm][f] = wmma_bf16(qa[mm][0], kf[f][0], s[mm][f]);
        s[mm][f] = wmma_bf16(qa[mm][1], kf[f][1], s[mm][f]);
      }

    // Online softmax (rows replicated across each 16-lane half).
#pragma unroll
    for (int mm = 0; mm < 2; ++mm) {
#pragma unroll
      for (int r = 0; r < 8; ++r) {
        const int qrow = q0 + mm * 16 + r + hf * 8;
        float sv0 = s[mm][0][r] * cscale;
        float sv1 = s[mm][1][r] * cscale;
        if (kb0 + lm > qrow)      sv0 = -__builtin_inff();
        if (kb0 + 16 + lm > qrow) sv1 = -__builtin_inff();
        const float rm = half_max16(fmaxf(sv0, sv1));
        const float mnew = fmaxf(mrun[mm][r], rm);
        const float alpha = __builtin_exp2f(mrun[mm][r] - mnew);
        const float p0 = __builtin_exp2f(sv0 - mnew);
        const float p1 = __builtin_exp2f(sv1 - mnew);
        lrun[mm][r] = lrun[mm][r] * alpha + half_sum16(p0 + p1);
        mrun[mm][r] = mnew;
#pragma unroll
        for (int f = 0; f < 4; ++f) o[mm][f][r] *= alpha;
        const int lrow = (mm * 16 + r + hf * 8) * 32;  // P row-major 32x32
        pl[lrow + lm] = (__bf16)p0;
        pl[lrow + 16 + lm] = (__bf16)p1;
      }
    }
    // C-layout -> A-layout relayout via private LDS tile; cross-lane dep, so
    // fence the DS counter (CDNA5 split wait) + compiler barrier.
    asm volatile("s_wait_dscnt 0x0" ::: "memory");
    v16bf pa[2];
    pa[0] = load_frag(pl, lm, 32, 0, hf);
    pa[1] = load_frag(pl + 16 * 32, lm, 32, 0, hf);
    asm volatile("" ::: "memory");  // keep next iter's DS stores after loads
#pragma unroll
    for (int mm = 0; mm < 2; ++mm)
#pragma unroll
      for (int f = 0; f < 4; ++f)
        o[mm][f] = wmma_bf16(pa[mm], vb[f], o[mm][f]);
  }

  const int bidx = bh >> 4, h = bh & 15;
#pragma unroll
  for (int mm = 0; mm < 2; ++mm)
#pragma unroll
    for (int f = 0; f < 4; ++f)
#pragma unroll
      for (int r = 0; r < 8; ++r) {
        const int l = q0 + mm * 16 + r + hf * 8;
        const float val = o[mm][f][r] / lrun[mm][r];
        AO[(size_t)(bidx * L_ + l) * D_ + h * DH_ + f * 16 + lm] = (__bf16)val;
      }
}

extern "C" void kernel_launch(void* const* d_in, const int* in_sizes, int n_in,
                              void* d_out, int out_size, void* d_ws,
                              size_t ws_size, hipStream_t stream) {
  (void)in_sizes; (void)n_in; (void)out_size; (void)ws_size;
  const float* x  = (const float*)d_in[0];
  // d_in[1] = causal mask (handled analytically)
  const float* wq = (const float*)d_in[2];
  const float* bq = (const float*)d_in[3];
  const float* wk = (const float*)d_in[4];
  const float* bk = (const float*)d_in[5];
  const float* wv = (const float*)d_in[6];
  const float* bv = (const float*)d_in[7];
  const float* wo = (const float*)d_in[8];
  const float* bo = (const float*)d_in[9];

  const size_t XN = (size_t)B_ * L_ * D_;  // 4194304
  const size_t WN = (size_t)D_ * D_;       // 1048576
  __bf16* ws   = (__bf16*)d_ws;            // ~50MB bf16 scratch
  __bf16* xb   = ws;
  __bf16* wqT  = xb + XN;
  __bf16* wkT  = wqT + WN;
  __bf16* wvT  = wkT + WN;
  __bf16* woT  = wvT + WN;
  __bf16* qws  = woT + WN;                 // [B,H,L,Dh]
  __bf16* kws  = qws + XN;                 // [B,H,L,Dh]
  __bf16* vws  = kws + XN;                 // [B,H,Dh,L] (transposed)
  __bf16* aows = vws + XN;                 // [B,L,D]

  cvt_bf16<<<(int)((XN + 255) / 256), 256, 0, stream>>>(x, xb, (int)XN);
  cvt_transpose<<<(int)((WN + 255) / 256), 256, 0, stream>>>(wq, wqT);
  cvt_transpose<<<(int)((WN + 255) / 256), 256, 0, stream>>>(wk, wkT);
  cvt_transpose<<<(int)((WN + 255) / 256), 256, 0, stream>>>(wv, wvT);
  cvt_transpose<<<(int)((WN + 255) / 256), 256, 0, stream>>>(wo, woT);

  dim3 ggrid(32, 16);  // (M/32/4, N/64) for M=4096, N=1024
  gemm_bf16<<<ggrid, 128, 0, stream>>>(xb, wqT, bq, qws, 0);
  gemm_bf16<<<ggrid, 128, 0, stream>>>(xb, wkT, bk, kws, 0);
  gemm_bf16<<<ggrid, 128, 0, stream>>>(xb, wvT, bv, vws, 2);

  flash_kernel<<<512, 128, 0, stream>>>(qws, kws, vws, aows);

  gemm_bf16<<<ggrid, 128, 0, stream>>>(aows, woT, bo, d_out, 3);
}
